// MyModelONNX_266287972380
// MI455X (gfx1250) — matrix operand order repaired and checked by reading, not compiled, not run on hardware
//
#include <hip/hip_runtime.h>
#include <hip/hip_bf16.h>

typedef _Float16 v16h __attribute__((ext_vector_type(16)));
typedef _Float16 h8   __attribute__((ext_vector_type(8)));
typedef _Float16 h4   __attribute__((ext_vector_type(4)));
typedef float    v8f  __attribute__((ext_vector_type(8)));

#define NUM_HEADS 4
#define SEQ 144
#define HDIM 32
#define SCALE 0.17677669529663687f

// LDS partition, in _Float16 units
#define Q_OFF   0                       // [4][144][32]
#define K_OFF   18432                   // [4][144][32]
#define VT_OFF  36864                   // [4][32][160]  (V transposed, m padded to 160)
#define P_OFF   57344                   // [8 waves][16][160]
#define SMEM_HALVES 77824
#define SMEM_BYTES (SMEM_HALVES * 2)    // 155648 B

union V16 { v16h v; h8 h[2]; };

__global__ void attn_fused_wmma(const float* __restrict__ qkv,
                                const float* __restrict__ mask,
                                float* __restrict__ out) {
    extern __shared__ __align__(16) _Float16 smem[];
    _Float16* q_s  = smem + Q_OFF;
    _Float16* k_s  = smem + K_OFF;
    _Float16* vt_s = smem + VT_OFF;
    _Float16* p_s  = smem + P_OFF;

    const int b   = blockIdx.x;
    const int tid = threadIdx.x;

    // ---------------- Phase 1: cooperative load + fp32 -> f16 convert ----------------
    const float4* src = reinterpret_cast<const float4*>(qkv + (size_t)b * SEQ * 384);
    for (int i = tid; i < (SEQ * 384 / 4); i += 256) {
        const int e  = i << 2;
        const int n  = e / 384;
        const int r  = e - n * 384;
        const int s  = r >> 7;          // 0=q 1=k 2=v
        const int hh = (r >> 5) & 3;    // head
        const int d  = r & 31;          // multiple of 4
        float4 f = src[i];
        if (s == 0) {
            h4 hv;
            hv.x = (_Float16)(f.x * SCALE); hv.y = (_Float16)(f.y * SCALE);
            hv.z = (_Float16)(f.z * SCALE); hv.w = (_Float16)(f.w * SCALE);
            *reinterpret_cast<h4*>(q_s + ((hh * SEQ + n) * HDIM + d)) = hv;
        } else if (s == 1) {
            h4 hv;
            hv.x = (_Float16)f.x; hv.y = (_Float16)f.y;
            hv.z = (_Float16)f.z; hv.w = (_Float16)f.w;
            *reinterpret_cast<h4*>(k_s + ((hh * SEQ + n) * HDIM + d)) = hv;
        } else {
            _Float16* base = vt_s + ((hh * HDIM + d) * 160 + n);   // transpose into [d][m]
            base[0]   = (_Float16)f.x;
            base[160] = (_Float16)f.y;
            base[320] = (_Float16)f.z;
            base[480] = (_Float16)f.w;
        }
    }
    // zero V^T pad columns m in [144,160)
    for (int i = tid; i < 2048; i += 256) {
        const int m    = 144 + (i & 15);
        const int rest = i >> 4;                    // h*32 + d, 0..127
        vt_s[rest * 160 + m] = (_Float16)0.0f;
    }
    // zero P pad columns (written once; compute only writes cols < 144)
    for (int i = tid; i < 2048; i += 256) {
        const int c   = 144 + (i & 15);
        const int row = (i >> 4) & 15;
        const int wv  = i >> 8;
        p_s[wv * 2560 + row * 160 + c] = (_Float16)0.0f;
    }
    __syncthreads();

    // ---------------- Phase 2: per-wave attention ----------------
    const int wave  = tid >> 5;
    const int lane  = tid & 31;
    const int l16   = lane & 15;
    const int lh    = lane >> 4;        // lane-half select
    const int h     = wave & 3;         // head for this wave
    const int whalf = wave >> 2;        // row-tile interleave

    const _Float16* qh  = q_s  + h * SEQ * HDIM;
    const _Float16* kh  = k_s  + h * SEQ * HDIM;
    const _Float16* vth = vt_s + h * HDIM * 160;
    _Float16*       pw  = p_s  + wave * 2560;

    for (int rt = whalf; rt < 9; rt += 2) {
        // A fragment (Q, 16x32 f16): lane-half K split {0-7,16-23} / {8-15,24-31}
        V16 aq;
        {
            const _Float16* qr = qh + (rt * 16 + l16) * HDIM;
            aq.h[0] = *reinterpret_cast<const h8*>(qr + lh * 8);
            aq.h[1] = *reinterpret_cast<const h8*>(qr + 16 + lh * 8);
        }

        // S = scale*q @ k^T + mask : 9 column tiles, one WMMA each (K == d == 32)
        v8f acc[9];
        #pragma unroll
        for (int ct = 0; ct < 9; ++ct) {
            v8f c;
            const float* mrow = mask + ((size_t)(h * SEQ + rt * 16 + 8 * lh)) * SEQ
                                     + ct * 16 + l16;
            #pragma unroll
            for (int r = 0; r < 8; ++r) c[r] = mrow[r * SEQ];
            V16 bk;   // B fragment (K^T, 32x16): col = l16, lane-half K split {0-7,16-23} / {8-15,24-31}
            const _Float16* kr = kh + (ct * 16 + l16) * HDIM + lh * 8;
            bk.h[0] = *reinterpret_cast<const h8*>(kr);
            bk.h[1] = *reinterpret_cast<const h8*>(kr + 16);
            acc[ct] = __builtin_amdgcn_wmma_f32_16x16x32_f16(
                false, aq.v, false, bk.v, (short)0, c, false, false);
        }

        // ---- softmax over 144 columns ----
        v8f rmax = acc[0];
        #pragma unroll
        for (int ct = 1; ct < 9; ++ct)
            #pragma unroll
            for (int r = 0; r < 8; ++r) rmax[r] = fmaxf(rmax[r], acc[ct][r]);
        #pragma unroll
        for (int r = 0; r < 8; ++r) {
            float m = rmax[r];
            m = fmaxf(m, __shfl_xor(m, 1, 16));
            m = fmaxf(m, __shfl_xor(m, 2, 16));
            m = fmaxf(m, __shfl_xor(m, 4, 16));
            m = fmaxf(m, __shfl_xor(m, 8, 16));
            rmax[r] = m;
        }
        v8f rsum;
        #pragma unroll
        for (int r = 0; r < 8; ++r) rsum[r] = 0.0f;
        #pragma unroll
        for (int ct = 0; ct < 9; ++ct) {
            #pragma unroll
            for (int r = 0; r < 8; ++r) {
                float p = __expf(acc[ct][r] - rmax[r]);
                rsum[r] += p;
                pw[(r + 8 * lh) * 160 + ct * 16 + l16] = (_Float16)p;
            }
        }
        #pragma unroll
        for (int r = 0; r < 8; ++r) {
            float s = rsum[r];
            s += __shfl_xor(s, 1, 16);
            s += __shfl_xor(s, 2, 16);
            s += __shfl_xor(s, 4, 16);
            s += __shfl_xor(s, 8, 16);
            rsum[r] = 1.0f / s;
        }

        // LDS store->load of P crosses lanes: order DS ops before fragment reads
        asm volatile("s_wait_dscnt 0" ::: "memory");

        // ---- O = P @ V : 2 output d-tiles, K padded to 160 (5 chunks of 32) ----
        #pragma unroll
        for (int dt = 0; dt < 2; ++dt) {
            v8f o;
            #pragma unroll
            for (int r = 0; r < 8; ++r) o[r] = 0.0f;
            #pragma unroll
            for (int c5 = 0; c5 < 5; ++c5) {
                V16 ap;   // A fragment from P: row = l16
                const _Float16* pr = pw + l16 * 160 + c5 * 32;
                ap.h[0] = *reinterpret_cast<const h8*>(pr + lh * 8);
                ap.h[1] = *reinterpret_cast<const h8*>(pr + 16 + lh * 8);
                V16 bv;   // B fragment from V^T: col(d) = dt*16+l16, m split {0-7,16-23} / {8-15,24-31}
                const _Float16* vr = vth + (dt * 16 + l16) * 160 + c5 * 32 + lh * 8;
                bv.h[0] = *reinterpret_cast<const h8*>(vr);
                bv.h[1] = *reinterpret_cast<const h8*>(vr + 16);
                o = __builtin_amdgcn_wmma_f32_16x16x32_f16(
                    false, ap.v, false, bv.v, (short)0, o, false, false);
            }
            float* orow = out + ((size_t)b * SEQ + rt * 16 + 8 * lh) * 128
                              + h * HDIM + dt * 16 + l16;
            #pragma unroll
            for (int r = 0; r < 8; ++r) orow[r * 128] = o[r] * rsum[r];
        }
    }
}

extern "C" void kernel_launch(void* const* d_in, const int* in_sizes, int n_in,
                              void* d_out, int out_size, void* d_ws, size_t ws_size,
                              hipStream_t stream) {
    (void)in_sizes; (void)n_in; (void)out_size; (void)d_ws; (void)ws_size;
    const float* qkv  = (const float*)d_in[0];
    const float* mask = (const float*)d_in[1];
    float* out = (float*)d_out;

    // allow >64KB dynamic LDS (gfx1250 WGP has 320KB)
    (void)hipFuncSetAttribute((const void*)attn_fused_wmma,
                              hipFuncAttributeMaxDynamicSharedMemorySize,
                              SMEM_BYTES);

    attn_fused_wmma<<<2048, 256, SMEM_BYTES, stream>>>(qkv, mask, out);
}
